// NonLocalBlock2d_3100966387739
// MI455X (gfx1250) — matrix operand order repaired and checked by reading, not compiled, not run on hardware
//
#include <hip/hip_runtime.h>

// ---------------------------------------------------------------------------
// NonLocalBlock2d for MI455X (gfx1250): bf16 WMMA GEMMs + flash attention.
// B=4, C=512, D=256, N=4096. wave32, V_WMMA_F32_16X16X32_BF16 everywhere.
// Round 2: double-buffered LDS pipelines (register staging) + prefetch,
//          branch-free k1 store epilogue.
// ---------------------------------------------------------------------------

typedef __attribute__((ext_vector_type(16))) __bf16 v16bf;
typedef __attribute__((ext_vector_type(8)))  float  v8f;

#define CB 4
#define CC 512
#define CD 256
#define CN 4096
#define ATT_SCALE 0.0625f  // D^-0.5 = 1/16
#define BN_EPS 1e-5f

static __device__ __forceinline__ v8f wmma_bf16(v16bf a, v16bf b, v8f c) {
  // D(16x16,f32) = A(16x32,bf16) x B(32x16,bf16) + C
  return __builtin_amdgcn_wmma_f32_16x16x32_bf16(
      /*neg_a=*/false, a, /*neg_b=*/false, b,
      /*c_mod=*/(short)0, c, /*reuse_a=*/false, /*reuse_b=*/false);
}

// A fragment (16x32, 16-bit): lane holds row (lane&15); kh = lane>>4.
// element e<8  -> K = 8*kh + e        (contiguous 8)
// element e>=8 -> K = 16 + 8*kh + e-8 (contiguous 8)
static __device__ __forceinline__ v16bf frag_a(const __bf16* p, int kh) {
  v16bf a;
  const __bf16* q0 = p + 8 * kh;
  const __bf16* q1 = p + 16 + 8 * kh;
#pragma unroll
  for (int i = 0; i < 8; ++i) { a[i] = q0[i]; a[8 + i] = q1[i]; }
  return a;
}

static __device__ __forceinline__ v16bf frag_a_f32(const float* p, int kh) {
  v16bf a;
  const float* q0 = p + 8 * kh;
  const float* q1 = p + 16 + 8 * kh;
#pragma unroll
  for (int i = 0; i < 8; ++i) { a[i] = (__bf16)q0[i]; a[8 + i] = (__bf16)q1[i]; }
  return a;
}

// B fragment (32x16, 16-bit): lane holds column (lane&15); e<8 -> K = 8*kh + e, e>=8 -> K = 16 + 8*kh + e-8.
static __device__ __forceinline__ v16bf frag_b(const __bf16* p, int kh) {
  v16bf b;
  const __bf16* q0 = p + 8 * kh;
  const __bf16* q1 = p + 16 + 8 * kh;
#pragma unroll
  for (int i = 0; i < 8; ++i) { b[i] = q0[i]; b[8 + i] = q1[i]; }
  return b;
}

static __device__ __forceinline__ float hmax16(float v) {
#pragma unroll
  for (int off = 1; off < 16; off <<= 1) v = fmaxf(v, __shfl_xor(v, off, 16));
  return v;
}
static __device__ __forceinline__ float hsum16(float v) {
#pragma unroll
  for (int off = 1; off < 16; off <<= 1) v += __shfl_xor(v, off, 16);
  return v;
}

// ---------------------------------------------------------------------------
// k0: zero batchnorm stat accumulators (1024 floats)
// ---------------------------------------------------------------------------
__global__ void k0_zero(float* p) {
  int i = blockIdx.x * blockDim.x + threadIdx.x;
  if (i < 2 * CC) p[i] = 0.0f;
}

// ---------------------------------------------------------------------------
// k1: fused theta/phi/g projections.  Stacked GEMM [768 x 512] @ [512 x N].
// Outputs: theta_t (B,N,D) bf16, phi_t (B,N,D) bf16, g (B,D,N) bf16.
// Workgroup: 256 thr (8 waves), tile M=128, N=64; wave = 32x32 (2x2 WMMA).
// Double-buffered LDS with register staging.
// ---------------------------------------------------------------------------
__global__ __launch_bounds__(256) void k1_qkv(
    const float* __restrict__ x,
    const float* __restrict__ w_theta, const float* __restrict__ b_theta,
    const float* __restrict__ w_phi,   const float* __restrict__ b_phi,
    const float* __restrict__ w_g,     const float* __restrict__ b_g,
    __bf16* __restrict__ theta_t, __bf16* __restrict__ phi_t,
    __bf16* __restrict__ gOut) {
  __shared__ __bf16 Wt[2 * 128 * 40];  // [buf][row][k], stride 40 (16B rows)
  __shared__ __bf16 Xt[2 * 64 * 40];   // [buf][n][k]   (transposed x tile)

  const int t    = threadIdx.x;
  const int lane = t & 31;
  const int wv   = t >> 5;
  const int l15  = lane & 15;
  const int kh   = lane >> 4;

  const int nBase  = blockIdx.x * 64;
  const int mBlock = blockIdx.y;        // 0..5 -> stacked rows of 128
  const int bb     = blockIdx.z;
  const int R0     = mBlock * 128;
  const int proj   = R0 >> 8;           // 0=theta 1=phi 2=g
  const int rOff   = R0 & 255;          // row offset within projection

  const float* wsrc = (proj == 0) ? w_theta : (proj == 1) ? w_phi : w_g;
  const float* bsrc = (proj == 0) ? b_theta : (proj == 1) ? b_phi : b_g;

  const int mW = (wv >> 1) * 32;
  const int nW = (wv & 1) * 32;

  v8f acc[2][2];
#pragma unroll
  for (int i = 0; i < 2; ++i)
#pragma unroll
    for (int j = 0; j < 2; ++j) acc[i][j] = (v8f){0, 0, 0, 0, 0, 0, 0, 0};

  // register staging for the software pipeline
  const int rW  = t >> 1;
  const int kW  = (t & 1) * 16;
  const int kkX = t & 31;
  const int ngX = t >> 5;
  float sW[16], sX[8];
  auto stage = [&](int kb) {
    const float* srcW = wsrc + (size_t)(rOff + rW) * CC + kb + kW;
#pragma unroll
    for (int i = 0; i < 16; ++i) sW[i] = srcW[i];
    const float* srcX = x + ((size_t)bb * CC + (kb + kkX)) * CN + nBase + ngX * 8;
#pragma unroll
    for (int i = 0; i < 8; ++i) sX[i] = srcX[i];
  };

  stage(0);
  for (int kb = 0; kb < CC; kb += 32) {
    const int buf = (kb >> 5) & 1;
    {  // drain staged registers into this iteration's LDS buffer
      __bf16* dW = Wt + buf * (128 * 40) + rW * 40 + kW;
#pragma unroll
      for (int i = 0; i < 16; ++i) dW[i] = (__bf16)sW[i];
      __bf16* dX = Xt + buf * (64 * 40);
#pragma unroll
      for (int i = 0; i < 8; ++i) dX[(ngX * 8 + i) * 40 + kkX] = (__bf16)sX[i];
    }
    __syncthreads();
    if (kb + 32 < CC) stage(kb + 32);  // loads for next tile in flight

    const __bf16* wB = Wt + buf * (128 * 40);
    const __bf16* xB = Xt + buf * (64 * 40);
    v16bf af[2], bf[2];
#pragma unroll
    for (int tm = 0; tm < 2; ++tm)
      af[tm] = frag_a(wB + (mW + tm * 16 + l15) * 40, kh);
#pragma unroll
    for (int tn = 0; tn < 2; ++tn)
      bf[tn] = frag_b(xB + (nW + tn * 16 + l15) * 40, kh);
#pragma unroll
    for (int tm = 0; tm < 2; ++tm)
#pragma unroll
      for (int tn = 0; tn < 2; ++tn)
        acc[tm][tn] = wmma_bf16(af[tm], bf[tn], acc[tm][tn]);
  }

  // bias + scatter stores, projection branch hoisted (uniform per block)
  if (proj < 2) {
    __bf16* dst = (proj == 0) ? theta_t : phi_t;
#pragma unroll
    for (int tm = 0; tm < 2; ++tm) {
#pragma unroll
      for (int r = 0; r < 8; ++r) {
        const int cc = rOff + mW + tm * 16 + r + 8 * kh;
        const float bias = bsrc[cc];
#pragma unroll
        for (int tn = 0; tn < 2; ++tn) {
          const int n = nBase + nW + tn * 16 + l15;
          dst[((size_t)bb * CN + n) * CD + cc] =
              (__bf16)(acc[tm][tn][r] + bias);
        }
      }
    }
  } else {
#pragma unroll
    for (int tm = 0; tm < 2; ++tm) {
#pragma unroll
      for (int r = 0; r < 8; ++r) {
        const int cc = rOff + mW + tm * 16 + r + 8 * kh;
        const float bias = bsrc[cc];
#pragma unroll
        for (int tn = 0; tn < 2; ++tn) {
          const int n = nBase + nW + tn * 16 + l15;
          gOut[((size_t)bb * CD + cc) * CN + n] =
              (__bf16)(acc[tm][tn][r] + bias);
        }
      }
    }
  }
}

// ---------------------------------------------------------------------------
// k2: flash attention.  Each workgroup: one batch, 128 query rows.
// Each wave: 16 rows; theta A-frags + 16 output d-tiles live in registers.
// Streams over key/value columns in steps of 32 with online softmax.
// Double-buffered LDS tiles with register staging + L2 prefetch.
// ---------------------------------------------------------------------------
__global__ __launch_bounds__(256) void k2_attn(
    const __bf16* __restrict__ theta_t, const __bf16* __restrict__ phi_t,
    const __bf16* __restrict__ gIn, __bf16* __restrict__ y_t) {
  __shared__ __bf16 phiS[2 * 32 * 264];   // [buf][m][d] keys tile
  __shared__ __bf16 gS[2 * 256 * 40];     // [buf][d][m] values tile
  __shared__ __bf16 pS[8][16 * 40];       // wave-private P transpose patch

  const int t    = threadIdx.x;
  const int lane = t & 31;
  const int wv   = t >> 5;
  const int l15  = lane & 15;
  const int kh   = lane >> 4;
  const int bb   = blockIdx.y;
  const int n0   = blockIdx.x * 128 + wv * 16;  // this wave's query rows

  // theta A fragments: rows fixed, K = d (8 steps of 32)
  v16bf aT[8];
  {
    const __bf16* trow = theta_t + ((size_t)bb * CN + n0 + l15) * CD;
#pragma unroll
    for (int kk = 0; kk < 8; ++kk) aT[kk] = frag_a(trow + 32 * kk, kh);
  }

  v8f acc[16];
#pragma unroll
  for (int i = 0; i < 16; ++i) acc[i] = (v8f){0, 0, 0, 0, 0, 0, 0, 0};
  float mrow[8], lrow[8];
#pragma unroll
  for (int r = 0; r < 8; ++r) { mrow[r] = -1e30f; lrow[r] = 0.0f; }

  // register staging for the tile pipeline (64B phi + 64B g per thread)
  const int mm = t >> 3;
  const int dc = (t & 7) * 32;
  const __bf16* phiBase = phi_t + (size_t)bb * CN * CD;
  const __bf16* gBase   = gIn + ((size_t)bb * CD + t) * CN;
  uint4 sP[4], sG[4];
  auto stage = [&](int m0) {
    const uint4* ps = (const uint4*)(phiBase + (size_t)(m0 + mm) * CD + dc);
    const uint4* gs = (const uint4*)(gBase + m0);
#pragma unroll
    for (int i = 0; i < 4; ++i) { sP[i] = ps[i]; sG[i] = gs[i]; }
  };

  stage(0);
  for (int m0 = 0; m0 < CN; m0 += 32) {
    const int buf = (m0 >> 5) & 1;
    {  // drain staged registers into this iteration's LDS buffers
      uint4* dp = (uint4*)(phiS + buf * (32 * 264) + mm * 264 + dc);
      uint4* dg = (uint4*)(gS + buf * (256 * 40) + t * 40);
#pragma unroll
      for (int i = 0; i < 4; ++i) { dp[i] = sP[i]; dg[i] = sG[i]; }
    }
    __syncthreads();
    if (m0 + 32 < CN) stage(m0 + 32);  // next tile's global loads in flight
    if (m0 + 64 < CN) {                // cover L2 latency for tile i+2
      __builtin_prefetch(phiBase + (size_t)(m0 + 64 + mm) * CD + dc, 0, 1);
      __builtin_prefetch(gBase + m0 + 64, 0, 1);
    }

    const __bf16* phiB = phiS + buf * (32 * 264);
    const __bf16* gB   = gS + buf * (256 * 40);

    // S = theta^T * phi over d (two 16-col tiles), K = 256
    v8f s0 = (v8f){0, 0, 0, 0, 0, 0, 0, 0};
    v8f s1 = (v8f){0, 0, 0, 0, 0, 0, 0, 0};
    const __bf16* pr0 = phiB + l15 * 264;
    const __bf16* pr1 = phiB + (16 + l15) * 264;
#pragma unroll
    for (int kk = 0; kk < 8; ++kk) {
      s0 = wmma_bf16(aT[kk], frag_b(pr0 + 32 * kk, kh), s0);
      s1 = wmma_bf16(aT[kk], frag_b(pr1 + 32 * kk, kh), s1);
    }

    // online softmax update (C/D layout: element r <-> row r + 8*kh)
    float alpha[8];
    __bf16* ps = &pS[wv][0];
#pragma unroll
    for (int r = 0; r < 8; ++r) {
      const float v0 = s0[r] * ATT_SCALE;
      const float v1 = s1[r] * ATT_SCALE;
      const float mx = hmax16(fmaxf(v0, v1));
      const float mn = fmaxf(mrow[r], mx);
      const float a  = __expf(mrow[r] - mn);
      const float e0 = __expf(v0 - mn);
      const float e1 = __expf(v1 - mn);
      lrow[r] = lrow[r] * a + hsum16(e0 + e1);
      mrow[r] = mn;
      alpha[r] = a;
      // stage P (D layout) for A-layout reload; wave-private, LDS in-order
      ps[(r + 8 * kh) * 40 + l15]      = (__bf16)e0;
      ps[(r + 8 * kh) * 40 + 16 + l15] = (__bf16)e1;
    }
#pragma unroll
    for (int tt = 0; tt < 16; ++tt)
#pragma unroll
      for (int r = 0; r < 8; ++r) acc[tt][r] *= alpha[r];

    const v16bf pf = frag_a(ps + l15 * 40, kh);
#pragma unroll
    for (int tt = 0; tt < 16; ++tt) {
      const v16bf bg = frag_b(gB + (16 * tt + l15) * 40, kh);
      acc[tt] = wmma_bf16(pf, bg, acc[tt]);
    }
  }

  // normalize and store y^T (B,N,D) bf16
  float inv[8];
#pragma unroll
  for (int r = 0; r < 8; ++r) inv[r] = 1.0f / lrow[r];
#pragma unroll
  for (int tt = 0; tt < 16; ++tt) {
#pragma unroll
    for (int r = 0; r < 8; ++r) {
      const int n = n0 + r + 8 * kh;
      const int d = 16 * tt + l15;
      y_t[((size_t)bb * CN + n) * CD + d] = (__bf16)(acc[tt][r] * inv[r]);
    }
  }
}

// ---------------------------------------------------------------------------
// k3: z = w_z @ y + b_z  (C=512 x N=4096, K=D=256) + per-channel BN stats.
// Workgroup tile: M=64, N=128 (8 waves, 32x32 each). Fragments straight from
// global (w_z rows and y_t rows are contiguous in K; everything L2-resident).
// ---------------------------------------------------------------------------
__global__ __launch_bounds__(256) void k3_zproj(
    const float* __restrict__ w_z, const float* __restrict__ b_z,
    const __bf16* __restrict__ y_t, float* __restrict__ z,
    float* __restrict__ sumB, float* __restrict__ sqB) {
  const int t    = threadIdx.x;
  const int lane = t & 31;
  const int wv   = t >> 5;
  const int l15  = lane & 15;
  const int kh   = lane >> 4;

  const int nBase = blockIdx.x * 128;
  const int cBase = blockIdx.y * 64;
  const int bb    = blockIdx.z;
  const int mW = (wv >> 2) * 32;
  const int nW = (wv & 3) * 32;

  v8f acc[2][2];
#pragma unroll
  for (int i = 0; i < 2; ++i)
#pragma unroll
    for (int j = 0; j < 2; ++j) acc[i][j] = (v8f){0, 0, 0, 0, 0, 0, 0, 0};

#pragma unroll 2
  for (int kk = 0; kk < 8; ++kk) {
    const int d0 = kk * 32;
    v16bf af[2], bf[2];
#pragma unroll
    for (int tm = 0; tm < 2; ++tm) {
      const int c = cBase + mW + tm * 16 + l15;
      af[tm] = frag_a_f32(w_z + (size_t)c * CD + d0, kh);
    }
#pragma unroll
    for (int tn = 0; tn < 2; ++tn) {
      const int n = nBase + nW + tn * 16 + l15;
      bf[tn] = frag_b(y_t + ((size_t)bb * CN + n) * CD + d0, kh);
    }
#pragma unroll
    for (int tm = 0; tm < 2; ++tm)
#pragma unroll
      for (int tn = 0; tn < 2; ++tn)
        acc[tm][tn] = wmma_bf16(af[tm], bf[tn], acc[tm][tn]);
  }

#pragma unroll
  for (int tm = 0; tm < 2; ++tm) {
#pragma unroll
    for (int tn = 0; tn < 2; ++tn) {
#pragma unroll
      for (int r = 0; r < 8; ++r) {
        const int c = cBase + mW + tm * 16 + r + 8 * kh;
        const int n = nBase + nW + tn * 16 + l15;
        const float v = acc[tm][tn][r] + b_z[c];
        z[((size_t)bb * CC + c) * CN + n] = v;
        const float rs = hsum16(v);
        const float rq = hsum16(v * v);
        if (l15 == 0) {
          atomicAdd(&sumB[c], rs);
          atomicAdd(&sqB[c], rq);
        }
      }
    }
  }
}

// ---------------------------------------------------------------------------
// k3b: fold stats into per-channel scale/shift
// ---------------------------------------------------------------------------
__global__ void k3b_finalize(const float* __restrict__ sumB,
                             const float* __restrict__ sqB,
                             const float* __restrict__ gamma,
                             const float* __restrict__ beta,
                             float* __restrict__ scl, float* __restrict__ sft) {
  const int c = threadIdx.x;
  const float invBN = 1.0f / (float)(CB * CN);
  const float mean = sumB[c] * invBN;
  const float var = sqB[c] * invBN - mean * mean;
  const float rstd = rsqrtf(var + BN_EPS);
  const float s = gamma[c] * rstd;
  scl[c] = s;
  sft[c] = beta[c] - mean * s;
}

// ---------------------------------------------------------------------------
// k4: out = z*scale[c] + shift[c] + x   (float4 vectorized)
// ---------------------------------------------------------------------------
__global__ __launch_bounds__(256) void k4_bn_residual(
    const float* __restrict__ z, const float* __restrict__ x,
    const float* __restrict__ scl, const float* __restrict__ sft,
    float* __restrict__ out) {
  const size_t i = (size_t)blockIdx.x * blockDim.x + threadIdx.x;
  const size_t base = i * 4;
  const int c = (int)((base >> 12) & (CC - 1));  // (base/4096) % 512
  const float s = scl[c], h = sft[c];
  const float4 zv = ((const float4*)z)[i];
  const float4 xv = ((const float4*)x)[i];
  float4 o;
  o.x = zv.x * s + h + xv.x;
  o.y = zv.y * s + h + xv.y;
  o.z = zv.z * s + h + xv.z;
  o.w = zv.w * s + h + xv.w;
  ((float4*)out)[i] = o;
}

// ---------------------------------------------------------------------------
extern "C" void kernel_launch(void* const* d_in, const int* in_sizes, int n_in,
                              void* d_out, int out_size, void* d_ws,
                              size_t ws_size, hipStream_t stream) {
  const float* x       = (const float*)d_in[0];
  const float* w_theta = (const float*)d_in[1];
  const float* b_theta = (const float*)d_in[2];
  const float* w_phi   = (const float*)d_in[3];
  const float* b_phi   = (const float*)d_in[4];
  const float* w_g     = (const float*)d_in[5];
  const float* b_g     = (const float*)d_in[6];
  const float* w_z     = (const float*)d_in[7];
  const float* b_z     = (const float*)d_in[8];
  const float* gamma   = (const float*)d_in[9];
  const float* beta    = (const float*)d_in[10];
  float* out = (float*)d_out;

  const size_t nd = (size_t)CB * CN * CD;  // 4M elements per bf16 tensor
  __bf16* thetaT = (__bf16*)d_ws;
  __bf16* phiT   = thetaT + nd;
  __bf16* gM     = phiT + nd;
  __bf16* yT     = gM + nd;
  float* zBuf = (float*)(yT + nd);
  float* sumB = zBuf + (size_t)CB * CC * CN;
  float* sqB  = sumB + CC;
  float* scl  = sqB + CC;
  float* sft  = scl + CC;

  k0_zero<<<4, 256, 0, stream>>>(sumB);
  k1_qkv<<<dim3(CN / 64, 6, CB), 256, 0, stream>>>(
      x, w_theta, b_theta, w_phi, b_phi, w_g, b_g, thetaT, phiT, gM);
  k2_attn<<<dim3(CN / 128, CB), 256, 0, stream>>>(thetaT, phiT, gM, yT);
  k3_zproj<<<dim3(CN / 128, CC / 64, CB), 256, 0, stream>>>(
      w_z, b_z, yT, zBuf, sumB, sqB);
  k3b_finalize<<<1, CC, 0, stream>>>(sumB, sqB, gamma, beta, scl, sft);
  k4_bn_residual<<<(CB * CC * CN / 4) / 256, 256, 0, stream>>>(
      zBuf, x, scl, sft, out);
}
